// ParticleNetOld_53712861003952
// MI455X (gfx1250) — matrix operand order repaired and checked by reading, not compiled, not run on hardware
//
#include <hip/hip_runtime.h>

// ---------------------------------------------------------------------------
// ParticleNet forward for MI455X (gfx1250, wave32, WMMA).
// All channel-mixing GEMMs use v_wmma_f32_16x16x32_f16 (f16 in, f32 acc).
// Each wave computes a 16x64 output strip: one A fragment is reused across
// four WMMAs per K-step (4x better matrix-op density than 16x16 tiling).
// BN statistics are fused into the GEMM epilogue (per-channel sum/sumsq).
// Problem sizes: B=128, P=256, K=16.
// ---------------------------------------------------------------------------

typedef __attribute__((ext_vector_type(16))) _Float16 v16h;
typedef __attribute__((ext_vector_type(8)))  _Float16 v8h;
typedef __attribute__((ext_vector_type(8)))  float    v8f;

#define BN_EPS 1e-5f

#define NB  128          // batch
#define NP  256          // points
#define NK  16           // neighbors
#define NBP (NB * NP)    // 32768
#define NBPK ((long)NB * NP * NK) // 524288

// ---------------------------------------------------------------------------
// WMMA GEMM:  Y[M,N] = A[M,Kd] (f16, row-major, lda) x W[N,Kd]^T
// (f16, row-major "out x in" = exactly the layout the reference weights use).
// Each wave owns a 16x64 strip (4 accumulators); N must be a multiple of 64.
// Optionally accumulates per-column sum / sum-of-squares into stats
// (sum at stats[n], sumsq at stats[256+n]) for fused batch-norm statistics.
// ---------------------------------------------------------------------------
__global__ void wmma_gemm_kernel(const _Float16* __restrict__ A, int lda,
                                 const _Float16* __restrict__ W, int ldw,
                                 float* __restrict__ Y, int ldy,
                                 long M, int N, int Kd,
                                 float* __restrict__ stats)
{
    const int wave = threadIdx.x >> 5;
    const int lane = threadIdx.x & 31;
    const int strips_n = N >> 6;               // 16x64 strips along N
    const long nstrips = (M >> 4) * (long)strips_n;
    const long strip = (long)blockIdx.x * 8 + wave;
    if (strip >= nstrips) return;              // whole wave exits together

    const int tm = (int)(strip / strips_n);
    const int tn = (int)(strip % strips_n);
    const int hi = lane >> 4;                  // 0 or 1 (lane group)
    const int lo = lane & 15;

    // A fragment: lane holds M = lo, K = kb + hi*16 + e
    const _Float16* arow = A + (long)(tm * 16 + lo) * lda + hi * 16;
    // B fragments from W[N,K]: lane holds N = tn*64 + j*16 + lo,
    // K = kb + hi*16 + e  -> one contiguous 32B v16h load per fragment
    const _Float16* wrow = W + (long)(tn * 64 + lo) * ldw + hi * 16;
    const long wstep = (long)16 * ldw;         // next 16-column group

    v8f acc0 = {}, acc1 = {}, acc2 = {}, acc3 = {};
    for (int kb = 0; kb < Kd; kb += 32) {
        if (kb + 32 < Kd) {                    // prefetch next K-slab
            __builtin_prefetch(arow + kb + 32, 0, 0);
            __builtin_prefetch(wrow + kb + 32, 0, 0);
        }
        union { v16h v; v8h h[2]; } a;
        a.v = *(const v16h*)(arow + kb);
        v16h b0 = *(const v16h*)(wrow + kb);
        v16h b1 = *(const v16h*)(wrow + wstep + kb);
        v16h b2 = *(const v16h*)(wrow + 2 * wstep + kb);
        v16h b3 = *(const v16h*)(wrow + 3 * wstep + kb);
        acc0 = __builtin_amdgcn_wmma_f32_16x16x32_f16(false, a.v, false, b0,
                                                      (short)0, acc0, false, false);
        acc1 = __builtin_amdgcn_wmma_f32_16x16x32_f16(false, a.v, false, b1,
                                                      (short)0, acc1, false, false);
        acc2 = __builtin_amdgcn_wmma_f32_16x16x32_f16(false, a.v, false, b2,
                                                      (short)0, acc2, false, false);
        acc3 = __builtin_amdgcn_wmma_f32_16x16x32_f16(false, a.v, false, b3,
                                                      (short)0, acc3, false, false);
    }

    // C/D layout: VGPR r -> row tm*16 + r + hi*8, column = strip base + lo
    float* ybase = Y + (long)(tm * 16 + hi * 8) * ldy + tn * 64 + lo;
    v8f accs[4] = {acc0, acc1, acc2, acc3};
#pragma unroll
    for (int j = 0; j < 4; ++j) {
        float s1 = 0.f, s2 = 0.f;
#pragma unroll
        for (int r = 0; r < 8; ++r) {
            float v = accs[j][r];
            ybase[(long)r * ldy + j * 16] = v;
            s1 += v;
            s2 += v * v;
        }
        if (stats) {
            const int n = tn * 64 + j * 16 + lo;  // 8 values share one column
            atomicAdd(&stats[n], s1);
            atomicAdd(&stats[256 + n], s2);
        }
    }
}

// ---------------------------------------------------------------------------
// Elementwise BN (from fused stats) + ReLU (+ optional mask), f32 -> f16/f32.
// ---------------------------------------------------------------------------
__global__ void bnrelu_kernel(const float* __restrict__ Y, int N, long M,
                              const float* __restrict__ stats, float inv_cnt,
                              const float* __restrict__ g,
                              const float* __restrict__ b,
                              _Float16* __restrict__ X16,
                              float* __restrict__ X32,
                              const float* __restrict__ maskf)
{
    long t = (long)blockIdx.x * blockDim.x + threadIdx.x;
    long total = M * (long)N;
    if (t >= total) return;
    int  c = (int)(t % N);
    long m = t / N;
    float mean = stats[c] * inv_cnt;
    float var  = stats[256 + c] * inv_cnt - mean * mean;
    float r    = rsqrtf(var + BN_EPS);
    float v    = (Y[t] - mean) * r * g[c] + b[c];
    v = fmaxf(v, 0.f);
    if (maskf) v *= maskf[m];
    if (X16) X16[t] = (_Float16)v;
    if (X32) X32[t] = v;
}

// ---------------------------------------------------------------------------
// Shortcut combine: out = relu(bn(SC) + AGG) * mask ; writes f32 and an f16
// copy into the fusion concat buffer at column offset col0 (ld = 192).
// ---------------------------------------------------------------------------
__global__ void combine_kernel(const float* __restrict__ SC,
                               const float* __restrict__ AGG, int C, long BPl,
                               const float* __restrict__ stats, float inv_cnt,
                               const float* __restrict__ g,
                               const float* __restrict__ b,
                               const float* __restrict__ maskf,
                               float* __restrict__ outF,
                               _Float16* __restrict__ out16, int ld16, int col0)
{
    long t = (long)blockIdx.x * blockDim.x + threadIdx.x;
    if (t >= BPl * (long)C) return;
    int  c = (int)(t % C);
    long m = t / C;
    float mean = stats[c] * inv_cnt;
    float var  = stats[256 + c] * inv_cnt - mean * mean;
    float r    = rsqrtf(var + BN_EPS);
    float v    = (SC[t] - mean) * r * g[c] + b[c] + AGG[t];
    v = fmaxf(v, 0.f) * maskf[m];
    outF[t] = v;
    out16[m * ld16 + col0 + c] = (_Float16)v;
}

// ---------------------------------------------------------------------------
// Input prep: mask, per-batch counts, shifted coords, transposed features,
// and fused BN statistics for the input-feature batch norm.
// ---------------------------------------------------------------------------
__global__ void prep_kernel(const float* __restrict__ points,
                            const float* __restrict__ features,
                            float* __restrict__ maskf,
                            float* __restrict__ counts,
                            float* __restrict__ ptsShift,
                            float* __restrict__ ftsraw,
                            float* __restrict__ statsFts)
{
    long t = (long)blockIdx.x * blockDim.x + threadIdx.x;
    if (t >= (long)NBP) return;
    int b = (int)(t / NP), p = (int)(t % NP);
    float f[16], s = 0.f;
#pragma unroll
    for (int c = 0; c < 16; ++c) {
        f[c] = features[((long)b * 16 + c) * NP + p];
        s += fabsf(f[c]);
    }
    float m = (s != 0.f) ? 1.f : 0.f;
    maskf[t] = m;
    atomicAdd(&counts[b], m);
    float shift = (1.f - m) * 1e9f;
#pragma unroll
    for (int d = 0; d < 2; ++d)
        ptsShift[t * 2 + d] = points[((long)b * 2 + d) * NP + p] * m + shift;
#pragma unroll
    for (int c = 0; c < 16; ++c) {
        float v = f[c] * m;
        ftsraw[t * 16 + c] = v;
        atomicAdd(&statsFts[c], v);
        atomicAdd(&statsFts[256 + c], v * v);
    }
}

// Apply input BN; write f32 activations and a zero-padded (16->32) f16 copy
// so the K=16 shortcut GEMM can run with Kd=32.
__global__ void bn_fts_kernel(const float* __restrict__ ftsraw,
                              const float* __restrict__ maskf,
                              const float* __restrict__ stats, float inv_cnt,
                              const float* __restrict__ g,
                              const float* __restrict__ b,
                              float* __restrict__ fts32,
                              _Float16* __restrict__ fts16p)
{
    long t = (long)blockIdx.x * blockDim.x + threadIdx.x;
    if (t >= (long)NBP) return;
    float m = maskf[t];
#pragma unroll
    for (int c = 0; c < 16; ++c) {
        float mean = stats[c] * inv_cnt;
        float var  = stats[256 + c] * inv_cnt - mean * mean;
        float v = (ftsraw[t * 16 + c] - mean) * rsqrtf(var + BN_EPS) * g[c] + b[c];
        v *= m;
        fts32[t * 16 + c]  = v;
        fts16p[t * 32 + c] = (_Float16)v;
    }
#pragma unroll
    for (int c = 16; c < 32; ++c) fts16p[t * 32 + c] = (_Float16)0.f;
}

// coords = src + (1-mask)*1e9 (for the second kNN in feature space)
__global__ void shift_kernel(const float* __restrict__ src,
                             const float* __restrict__ maskf, int C,
                             float* __restrict__ dst, long total)
{
    long t = (long)blockIdx.x * blockDim.x + threadIdx.x;
    if (t >= total) return;
    dst[t] = src[t] + (1.f - maskf[t / C]) * 1e9f;
}

// ---------------------------------------------------------------------------
// kNN: per (b,p) keep the 17 smallest squared distances (insertion sort,
// stable for ties like lax.top_k), drop self, emit 16 neighbor indices.
// ---------------------------------------------------------------------------
template <int C>
__global__ void knn_kernel(const float* __restrict__ coords,
                           int* __restrict__ idx)
{
    long t = (long)blockIdx.x * blockDim.x + threadIdx.x;
    if (t >= (long)NBP) return;
    int b = (int)(t / NP), p = (int)(t % NP);
    const float* base = coords + (long)b * NP * C;
    float my[C];
#pragma unroll
    for (int c = 0; c < C; ++c) my[c] = base[(long)p * C + c];

    const int KK = NK + 1;
    float bd[KK];
    int   bi[KK];
#pragma unroll
    for (int i = 0; i < KK; ++i) { bd[i] = 3.4e38f; bi[i] = 0; }

    for (int q = 0; q < NP; ++q) {
        float d = 0.f;
#pragma unroll 4
        for (int c = 0; c < C; ++c) {
            float df = base[(long)q * C + c] - my[c];
            d += df * df;
        }
        if (d < bd[KK - 1]) {
            int j = KK - 1;
            while (j > 0 && d < bd[j - 1]) {
                bd[j] = bd[j - 1]; bi[j] = bi[j - 1]; --j;
            }
            bd[j] = d; bi[j] = q;
        }
    }
#pragma unroll
    for (int j = 0; j < NK; ++j) idx[t * NK + j] = bi[j + 1];
}

// Build edge features [ctr, nb - ctr] in f16 for the first conv of a block.
__global__ void gather_kernel(const float* __restrict__ F, int Cin,
                              const int* __restrict__ idx,
                              _Float16* __restrict__ X0, int ldx)
{
    long r = (long)blockIdx.x * blockDim.x + threadIdx.x;
    if (r >= NBPK) return;
    long bp = r / NK;
    long b  = bp / NP;
    int  nb = idx[r];
    const float* ctr = F + bp * Cin;
    const float* nbp = F + (b * NP + nb) * (long)Cin;
    _Float16* dst = X0 + r * ldx;
    for (int c = 0; c < Cin; ++c) {
        float cc = ctr[c];
        dst[c]       = (_Float16)cc;
        dst[Cin + c] = (_Float16)(nbp[c] - cc);
    }
}

// agg[bp,c] = mean_k X[(bp*K+k), c]
__global__ void meank_kernel(const _Float16* __restrict__ X, int C,
                             float* __restrict__ agg)
{
    long t = (long)blockIdx.x * blockDim.x + threadIdx.x;
    if (t >= (long)NBP * C) return;
    int  c  = (int)(t % C);
    long bp = t / C;
    float s = 0.f;
#pragma unroll
    for (int k = 0; k < NK; ++k)
        s += (float)X[(bp * NK + k) * (long)C + c];
    agg[t] = s * (1.f / NK);
}

// pooled[b,c] = sum_p fused[b,p,c] / max(counts[b],1)
__global__ void pool_kernel(const float* __restrict__ fused,
                            const float* __restrict__ counts, int C,
                            float* __restrict__ pooled,
                            _Float16* __restrict__ pooled16)
{
    long t = (long)blockIdx.x * blockDim.x + threadIdx.x;
    if (t >= (long)NB * C) return;
    int b = (int)(t / C), c = (int)(t % C);
    float s = 0.f;
    for (int p = 0; p < NP; ++p)
        s += fused[((long)b * NP + p) * C + c];
    float v = s / fmaxf(counts[b], 1.f);
    pooled[t]   = v;
    pooled16[t] = (_Float16)v;
}

__global__ void biasrelu_kernel(const float* __restrict__ Y,
                                const float* __restrict__ bias, int N,
                                float* __restrict__ out, long total)
{
    long t = (long)blockIdx.x * blockDim.x + threadIdx.x;
    if (t >= total) return;
    out[t] = fmaxf(Y[t] + bias[t % N], 0.f);
}

// Tiny final FC (N=10 not a multiple of 16 -> scalar f32).
__global__ void fc2_kernel(const float* __restrict__ x,
                           const float* __restrict__ w,
                           const float* __restrict__ b, int C, int O,
                           float* __restrict__ out)
{
    long t = (long)blockIdx.x * blockDim.x + threadIdx.x;
    if (t >= (long)NB * O) return;
    int bb = (int)(t / O), o = (int)(t % O);
    float s = b[o];
    for (int c = 0; c < C; ++c)
        s += x[(long)bb * C + c] * w[(long)o * C + c];
    out[t] = s;
}

// f32 -> f16 weight conversion with optional K padding (K=16 shortcut -> 32).
__global__ void cvt16_kernel(const float* __restrict__ src,
                             _Float16* __restrict__ dst, int N, int Ks, int Kp)
{
    long t = (long)blockIdx.x * blockDim.x + threadIdx.x;
    if (t >= (long)N * Kp) return;
    int c = (int)(t % Kp), n = (int)(t / Kp);
    dst[t] = (_Float16)((c < Ks) ? src[(long)n * Ks + c] : 0.f);
}

// ---------------------------------------------------------------------------
// Host side
// ---------------------------------------------------------------------------
static inline int cdiv(long a, int b) { return (int)((a + b - 1) / b); }

static void launch_gemm(const _Float16* A, int lda, const _Float16* W, int ldw,
                        float* Y, int ldy, long M, int N, int Kd,
                        float* stats, hipStream_t s)
{
    long nstrips = (M / 16) * (long)(N / 64);   // 16x64 strips per wave
    int blocks = (int)((nstrips + 7) / 8);
    hipLaunchKernelGGL(wmma_gemm_kernel, dim3(blocks), dim3(256), 0, s,
                       A, lda, W, ldw, Y, ldy, M, N, Kd, stats);
}

extern "C" void kernel_launch(void* const* d_in, const int* in_sizes, int n_in,
                              void* d_out, int out_size, void* d_ws,
                              size_t ws_size, hipStream_t stream)
{
    (void)in_sizes; (void)n_in; (void)out_size; (void)ws_size;

    // Input order: JAX pytree flatten (sorted dict keys).
    const float* features = (const float*)d_in[0];
    // block 0
    const float* b0_bnb[3] = {(const float*)d_in[1], (const float*)d_in[2], (const float*)d_in[3]};
    const float* b0_bng[3] = {(const float*)d_in[4], (const float*)d_in[5], (const float*)d_in[6]};
    const float* b0_cw[3]  = {(const float*)d_in[7], (const float*)d_in[8], (const float*)d_in[9]};
    const float* b0_scb = (const float*)d_in[10];
    const float* b0_scg = (const float*)d_in[11];
    const float* b0_scw = (const float*)d_in[12];
    // block 1
    const float* b1_bnb[3] = {(const float*)d_in[13], (const float*)d_in[14], (const float*)d_in[15]};
    const float* b1_bng[3] = {(const float*)d_in[16], (const float*)d_in[17], (const float*)d_in[18]};
    const float* b1_cw[3]  = {(const float*)d_in[19], (const float*)d_in[20], (const float*)d_in[21]};
    const float* b1_scb = (const float*)d_in[22];
    const float* b1_scg = (const float*)d_in[23];
    const float* b1_scw = (const float*)d_in[24];
    const float* bn_fts_b = (const float*)d_in[25];
    const float* bn_fts_g = (const float*)d_in[26];
    const float* fc1_b = (const float*)d_in[27];
    const float* fc1_w = (const float*)d_in[28];
    const float* fc2_b = (const float*)d_in[29];
    const float* fc2_w = (const float*)d_in[30];
    const float* fus_b = (const float*)d_in[31];
    const float* fus_g = (const float*)d_in[32];
    const float* fus_w = (const float*)d_in[33];
    const float* points = (const float*)d_in[34];

    // ---- bump allocator over d_ws ----
    char* base = (char*)d_ws;
    size_t off = 0;
    auto alloc = [&](size_t bytes) -> void* {
        void* p = base + off;
        off = (off + bytes + 255) & ~(size_t)255;
        return p;
    };

    float*    stats   = (float*)alloc(10 * 512 * sizeof(float)); // 10 slots
    float*    counts  = (float*)alloc(NB * sizeof(float));
    float*    maskf   = (float*)alloc(NBP * sizeof(float));
    float*    ptsSh   = (float*)alloc((size_t)NBP * 2 * sizeof(float));
    float*    ftsraw  = (float*)alloc((size_t)NBP * 16 * sizeof(float));
    float*    fts32   = (float*)alloc((size_t)NBP * 16 * sizeof(float));
    _Float16* fts16p  = (_Float16*)alloc((size_t)NBP * 32 * sizeof(_Float16));
    int*      idx     = (int*)alloc((size_t)NBPK * sizeof(int));
    _Float16* XA      = (_Float16*)alloc((size_t)NBPK * 128 * sizeof(_Float16));
    _Float16* XB      = (_Float16*)alloc((size_t)NBPK * 128 * sizeof(_Float16));
    float*    Yb      = (float*)alloc((size_t)NBPK * 128 * sizeof(float));
    float*    agg     = (float*)alloc((size_t)NBP * 128 * sizeof(float));
    float*    scY     = (float*)alloc((size_t)NBP * 128 * sizeof(float));
    float*    out0f   = (float*)alloc((size_t)NBP * 64 * sizeof(float));
    float*    coords1 = (float*)alloc((size_t)NBP * 64 * sizeof(float));
    float*    out1f   = (float*)alloc((size_t)NBP * 128 * sizeof(float));
    _Float16* cat16   = (_Float16*)alloc((size_t)NBP * 192 * sizeof(_Float16));
    float*    fusedf  = (float*)alloc((size_t)NBP * 128 * sizeof(float));
    float*    pooled  = (float*)alloc((size_t)NB * 128 * sizeof(float));
    _Float16* pooled16= (_Float16*)alloc((size_t)NB * 128 * sizeof(_Float16));
    float*    fc1Y    = (float*)alloc((size_t)NB * 256 * sizeof(float));
    float*    fc1a    = (float*)alloc((size_t)NB * 256 * sizeof(float));
    _Float16* w0c0 = (_Float16*)alloc(64 * 32 * sizeof(_Float16));
    _Float16* w0c1 = (_Float16*)alloc(64 * 64 * sizeof(_Float16));
    _Float16* w0c2 = (_Float16*)alloc(64 * 64 * sizeof(_Float16));
    _Float16* w0sc = (_Float16*)alloc(64 * 32 * sizeof(_Float16)); // 16 padded->32
    _Float16* w1c0 = (_Float16*)alloc(128 * 128 * sizeof(_Float16));
    _Float16* w1c1 = (_Float16*)alloc(128 * 128 * sizeof(_Float16));
    _Float16* w1c2 = (_Float16*)alloc(128 * 128 * sizeof(_Float16));
    _Float16* w1sc = (_Float16*)alloc(128 * 64 * sizeof(_Float16));
    _Float16* wfus = (_Float16*)alloc(128 * 192 * sizeof(_Float16));
    _Float16* wfc1 = (_Float16*)alloc(256 * 128 * sizeof(_Float16));

    // zero stat accumulators + counts (graph-capture-safe stream memsets)
    hipMemsetAsync(stats, 0, 10 * 512 * sizeof(float), stream);
    hipMemsetAsync(counts, 0, NB * sizeof(float), stream);

    const int T = 256;
    // ---- weight conversion ----
    auto cvt = [&](const float* s, _Float16* d, int N, int Ks, int Kp) {
        hipLaunchKernelGGL(cvt16_kernel, dim3(cdiv((long)N * Kp, T)), dim3(T),
                           0, stream, s, d, N, Ks, Kp);
    };
    cvt(b0_cw[0], w0c0, 64, 32, 32);
    cvt(b0_cw[1], w0c1, 64, 64, 64);
    cvt(b0_cw[2], w0c2, 64, 64, 64);
    cvt(b0_scw,  w0sc, 64, 16, 32);
    cvt(b1_cw[0], w1c0, 128, 128, 128);
    cvt(b1_cw[1], w1c1, 128, 128, 128);
    cvt(b1_cw[2], w1c2, 128, 128, 128);
    cvt(b1_scw,  w1sc, 128, 64, 64);
    cvt(fus_w,   wfus, 128, 192, 192);
    cvt(fc1_w,   wfc1, 256, 128, 128);

    const float invBP  = 1.f / (float)NBP;
    const float invBPK = 1.f / (float)NBPK;

    // ---- prep + input BN ----
    hipLaunchKernelGGL(prep_kernel, dim3(cdiv(NBP, T)), dim3(T), 0, stream,
                       points, features, maskf, counts, ptsSh, ftsraw,
                       stats + 0 * 512);
    hipLaunchKernelGGL(bn_fts_kernel, dim3(cdiv(NBP, T)), dim3(T), 0, stream,
                       ftsraw, maskf, stats + 0 * 512, invBP,
                       bn_fts_g, bn_fts_b, fts32, fts16p);

    // =================== EdgeConv block 0 (k=16, 64/64/64) ===================
    knn_kernel<2><<<cdiv(NBP, T), T, 0, stream>>>(ptsSh, idx);
    hipLaunchKernelGGL(gather_kernel, dim3(cdiv(NBPK, T)), dim3(T), 0, stream,
                       fts32, 16, idx, XA, 32);
    // conv 1..3 (GEMM + fused stats, then BN+ReLU -> f16)
    launch_gemm(XA, 32, w0c0, 32, Yb, 64, NBPK, 64, 32, stats + 1 * 512, stream);
    hipLaunchKernelGGL(bnrelu_kernel, dim3(cdiv(NBPK * 64, T)), dim3(T), 0, stream,
                       Yb, 64, NBPK, stats + 1 * 512, invBPK,
                       b0_bng[0], b0_bnb[0], XB, (float*)nullptr, (const float*)nullptr);
    launch_gemm(XB, 64, w0c1, 64, Yb, 64, NBPK, 64, 64, stats + 2 * 512, stream);
    hipLaunchKernelGGL(bnrelu_kernel, dim3(cdiv(NBPK * 64, T)), dim3(T), 0, stream,
                       Yb, 64, NBPK, stats + 2 * 512, invBPK,
                       b0_bng[1], b0_bnb[1], XA, (float*)nullptr, (const float*)nullptr);
    launch_gemm(XA, 64, w0c2, 64, Yb, 64, NBPK, 64, 64, stats + 3 * 512, stream);
    hipLaunchKernelGGL(bnrelu_kernel, dim3(cdiv(NBPK * 64, T)), dim3(T), 0, stream,
                       Yb, 64, NBPK, stats + 3 * 512, invBPK,
                       b0_bng[2], b0_bnb[2], XB, (float*)nullptr, (const float*)nullptr);
    hipLaunchKernelGGL(meank_kernel, dim3(cdiv((long)NBP * 64, T)), dim3(T), 0,
                       stream, XB, 64, agg);
    // shortcut (Kd padded 16->32) + combine
    launch_gemm(fts16p, 32, w0sc, 32, scY, 64, NBP, 64, 32, stats + 4 * 512, stream);
    hipLaunchKernelGGL(combine_kernel, dim3(cdiv((long)NBP * 64, T)), dim3(T), 0,
                       stream, scY, agg, 64, (long)NBP, stats + 4 * 512, invBP,
                       b0_scg, b0_scb, maskf, out0f, cat16, 192, 0);

    // =================== EdgeConv block 1 (k=16, 128/128/128) ================
    hipLaunchKernelGGL(shift_kernel, dim3(cdiv((long)NBP * 64, T)), dim3(T), 0,
                       stream, out0f, maskf, 64, coords1, (long)NBP * 64);
    knn_kernel<64><<<cdiv(NBP, T), T, 0, stream>>>(coords1, idx);
    hipLaunchKernelGGL(gather_kernel, dim3(cdiv(NBPK, T)), dim3(T), 0, stream,
                       out0f, 64, idx, XA, 128);
    launch_gemm(XA, 128, w1c0, 128, Yb, 128, NBPK, 128, 128, stats + 5 * 512, stream);
    hipLaunchKernelGGL(bnrelu_kernel, dim3(cdiv(NBPK * 128, T)), dim3(T), 0, stream,
                       Yb, 128, NBPK, stats + 5 * 512, invBPK,
                       b1_bng[0], b1_bnb[0], XB, (float*)nullptr, (const float*)nullptr);
    launch_gemm(XB, 128, w1c1, 128, Yb, 128, NBPK, 128, 128, stats + 6 * 512, stream);
    hipLaunchKernelGGL(bnrelu_kernel, dim3(cdiv(NBPK * 128, T)), dim3(T), 0, stream,
                       Yb, 128, NBPK, stats + 6 * 512, invBPK,
                       b1_bng[1], b1_bnb[1], XA, (float*)nullptr, (const float*)nullptr);
    launch_gemm(XA, 128, w1c2, 128, Yb, 128, NBPK, 128, 128, stats + 7 * 512, stream);
    hipLaunchKernelGGL(bnrelu_kernel, dim3(cdiv(NBPK * 128, T)), dim3(T), 0, stream,
                       Yb, 128, NBPK, stats + 7 * 512, invBPK,
                       b1_bng[2], b1_bnb[2], XB, (float*)nullptr, (const float*)nullptr);
    hipLaunchKernelGGL(meank_kernel, dim3(cdiv((long)NBP * 128, T)), dim3(T), 0,
                       stream, XB, 128, agg);
    // shortcut: A = block0 output living in cat16 cols [0,64), lda=192
    launch_gemm(cat16, 192, w1sc, 64, scY, 128, NBP, 128, 64, stats + 8 * 512, stream);
    hipLaunchKernelGGL(combine_kernel, dim3(cdiv((long)NBP * 128, T)), dim3(T), 0,
                       stream, scY, agg, 128, (long)NBP, stats + 8 * 512, invBP,
                       b1_scg, b1_scb, maskf, out1f, cat16, 192, 64);

    // =================== fusion + pool + FCs ===================
    launch_gemm(cat16, 192, wfus, 192, Yb, 128, NBP, 128, 192, stats + 9 * 512, stream);
    hipLaunchKernelGGL(bnrelu_kernel, dim3(cdiv((long)NBP * 128, T)), dim3(T), 0,
                       stream, Yb, 128, (long)NBP, stats + 9 * 512, invBP,
                       fus_g, fus_b, (_Float16*)nullptr, fusedf, maskf);
    hipLaunchKernelGGL(pool_kernel, dim3(cdiv((long)NB * 128, T)), dim3(T), 0,
                       stream, fusedf, counts, 128, pooled, pooled16);
    launch_gemm(pooled16, 128, wfc1, 128, fc1Y, 256, NB, 256, 128, nullptr, stream);
    hipLaunchKernelGGL(biasrelu_kernel, dim3(cdiv((long)NB * 256, T)), dim3(T), 0,
                       stream, fc1Y, fc1_b, 256, fc1a, (long)NB * 256);
    hipLaunchKernelGGL(fc2_kernel, dim3(cdiv((long)NB * 10, T)), dim3(T), 0,
                       stream, fc1a, fc2_w, fc2_b, 256, 10, (float*)d_out);
}
